// InnerVoice_73899207295210
// MI455X (gfx1250) — matrix operand order repaired and checked by reading, not compiled, not run on hardware
//
#include <hip/hip_runtime.h>
#include <hip/hip_bf16.h>

// ---------------- problem constants ----------------
constexpr int B_ = 16;
constexpr int T_ = 256;
constexpr int E_ = 512;
constexpr int H_ = 2048;
constexpr int V_ = 32000;
constexpr float DECAY = 0.95122942450071400910f;   // exp(-1/20)
constexpr float NOISE_STD = 0.02f;

// ---------------- WMMA vector types ----------------
typedef __attribute__((ext_vector_type(16))) __bf16 v16bf;
typedef __attribute__((ext_vector_type(8)))  __bf16 v8bf;
typedef __attribute__((ext_vector_type(8)))  float  v8f;
typedef __attribute__((ext_vector_type(4)))  float  v4f;

union BVec { v16bf v; v8bf h[2]; };

// B tile stored pre-swizzled: tile = 32 lanes x 16 bf16, contiguous.
// lane<16 holds K=0..7,16..23 (elem j == K (j&7)+16*(j>>3)) of column n=lane;
// lane>=16 holds K=8..15,24..31 of column n=lane-16.        (ISA 7.12.2)
__device__ __forceinline__ v16bf load_b_sw(const __bf16* bsw, long long tileIdx, int lane) {
    const __bf16* p = bsw + (tileIdx * 32 + lane) * 16;
    BVec b;
    b.h[0] = *(const v8bf*)(p);
    b.h[1] = *(const v8bf*)(p + 8);
    return b.v;
}

// A 16x32 bf16 layout: lane L, row M=L&15; lane<16: elems 0-7 = K k0..k0+7,
// elems 8-15 = K k0+16..k0+23 with k0 = kt*32; lane>=16: k0 = kt*32+8.
__device__ __forceinline__ v16bf load_a_bf16(const __bf16* arow, int k0) {
    BVec a;
    a.h[0] = *(const v8bf*)(arow + k0);
    a.h[1] = *(const v8bf*)(arow + k0 + 16);
    return a.v;
}

__device__ __forceinline__ v16bf load_a_f32(const float* arow, int k0) {
    v4f f0 = *(const v4f*)(arow + k0);
    v4f f1 = *(const v4f*)(arow + k0 + 4);
    v4f f2 = *(const v4f*)(arow + k0 + 16);
    v4f f3 = *(const v4f*)(arow + k0 + 20);
    v16bf a;
#pragma unroll
    for (int i = 0; i < 4; ++i) {
        a[i]      = (__bf16)f0[i];
        a[4 + i]  = (__bf16)f1[i];
        a[8 + i]  = (__bf16)f2[i];
        a[12 + i] = (__bf16)f3[i];
    }
    return a;
}

__device__ __forceinline__ v8f wmma_bf16(v16bf a, v16bf b, v8f c) {
    return __builtin_amdgcn_wmma_f32_16x16x32_bf16(false, a, false, b, (short)0, c, false, false);
}

// ---------------- prep kernels ----------------

__global__ void zero_state_kernel(float* __restrict__ mem, __bf16* __restrict__ hA) {
    int i = blockIdx.x * blockDim.x + threadIdx.x;
    if (i < B_ * H_) { mem[i] = 0.0f; hA[i] = (__bf16)0.0f; }
}

// fp32 [K][N] row-major  ->  bf16 WMMA-B swizzled tiles (tile order: nt-major, kt inner)
__global__ void swizzleB_kernel(const float* __restrict__ src, __bf16* __restrict__ dst,
                                int K, int N, int maskDiag) {
    long long idx = (long long)blockIdx.x * blockDim.x + threadIdx.x;
    long long total = (long long)K * N;
    if (idx >= total) return;
    int j    = (int)(idx & 15);
    int lane = (int)((idx >> 4) & 31);
    long long tile = idx >> 9;
    int KT = K >> 5;
    int kt = (int)(tile % KT);
    long long nt = tile / KT;
    int k = kt * 32 + ((lane >> 4) << 3) + (j & 7) + ((j >> 3) << 4);
    long long n = nt * 16 + (lane & 15);
    float v = src[(long long)k * N + n];
    if (maskDiag && (long long)k == n) v = 0.0f;   // recurrent_mask = 1 - I
    dst[idx] = (__bf16)v;
}

// ---------------- GEMM 1: inp = x @ W_in  (f32 A, swizzled bf16 B, f32 out) ----------------
__global__ void gemm_input_kernel(const float* __restrict__ x, const __bf16* __restrict__ winsw,
                                  float* __restrict__ inp) {
    constexpr int NT = H_ / 16;      // 128
    constexpr int KT = E_ / 32;      // 16
    int wave = blockIdx.x * (blockDim.x >> 5) + (threadIdx.x >> 5);
    int lane = threadIdx.x & 31;
    int mt = wave / NT, nt = wave % NT;
    int hi = lane >> 4;
    const float* arow = x + (long long)(mt * 16 + (lane & 15)) * E_;
    v8f c = {};
#pragma unroll 4
    for (int kt = 0; kt < KT; ++kt) {
        v16bf a = load_a_f32(arow, kt * 32 + hi * 8);
        v16bf b = load_b_sw(winsw, (long long)nt * KT + kt, lane);
        c = wmma_bf16(a, b, c);
    }
    int n = nt * 16 + (lane & 15);
#pragma unroll
    for (int r = 0; r < 8; ++r)
        inp[(long long)(mt * 16 + r + hi * 8) * H_ + n] = c[r];
}

// ---------------- GEMM 2 + LIF step (one timestep; launched T times) ----------------
__global__ void lif_step_kernel(const __bf16* __restrict__ hprev, const __bf16* __restrict__ wrecsw,
                                const float* __restrict__ inp, const float* __restrict__ noise,
                                float* __restrict__ mem, __bf16* __restrict__ hnext,
                                __bf16* __restrict__ hids, int t) {
    constexpr int KT = H_ / 32;      // 64
    int wave = blockIdx.x * (blockDim.x >> 5) + (threadIdx.x >> 5);  // nt = 0..127
    int lane = threadIdx.x & 31;
    int nt = wave;
    int hi = lane >> 4;
    const __bf16* arow = hprev + (long long)(lane & 15) * H_;   // row = batch index
    v8f c = {};
#pragma unroll 4
    for (int kt = 0; kt < KT; ++kt) {
        v16bf a = load_a_bf16(arow, kt * 32 + hi * 8);
        v16bf b = load_b_sw(wrecsw, (long long)nt * KT + kt, lane);
        c = wmma_bf16(a, b, c);
    }
    int n = nt * 16 + (lane & 15);   // hidden unit index
#pragma unroll
    for (int r = 0; r < 8; ++r) {
        int bb = r + hi * 8;                         // batch index
        long long bt = (long long)bb * T_ + t;
        float hin = c[r] + inp[bt * H_ + n];
        float mv  = DECAY * mem[(long long)bb * H_ + n] + hin + NOISE_STD * noise[bt * H_ + n];
        if (mv >= 1.0f) mv = 0.0f;                   // fire & reset
        float hd = 1.0f / (1.0f + __expf(-mv));      // hidden = sigmoid(mem)
        mem[(long long)bb * H_ + n] = mv;
        __bf16 hb = (__bf16)hd;
        hnext[(long long)bb * H_ + n] = hb;
        hids[((long long)t * B_ + bb) * H_ + n] = hb;   // [T,B,H] row-major
    }
}

// ---------------- GEMM 3: logits = hids @ W_out + bias ----------------
// Variant A: pre-swizzled bf16 W_out, register-blocked 2 M-tiles x 4 N-tiles per wave.
// 8 wmmas per K-step for 2 A-tile + 4 B-tile loads -> matrix-pipe bound, not load bound.
__global__ void gemm_out_sw_kernel(const __bf16* __restrict__ hids, const __bf16* __restrict__ woutsw,
                                   const float* __restrict__ bias, float* __restrict__ out) {
    constexpr int KT = H_ / 32;          // 64
    constexpr int NB = V_ / 64;          // 500 blocks of 4 N-tiles
    int wave = blockIdx.x * (blockDim.x >> 5) + (threadIdx.x >> 5);
    int lane = threadIdx.x & 31;
    int mb = wave / NB;                  // block of 2 timesteps (each M-tile = one t)
    int nb = wave % NB;
    int hi = lane >> 4;
    const __bf16* arow0 = hids + (long long)(mb * 32 + (lane & 15)) * H_;
    const __bf16* arow1 = arow0 + (long long)16 * H_;
    long long btile0 = (long long)(nb * 4) * KT;

    v8f c[2][4];
#pragma unroll
    for (int m = 0; m < 2; ++m)
#pragma unroll
        for (int i = 0; i < 4; ++i) c[m][i] = (v8f){};

#pragma unroll 2
    for (int kt = 0; kt < KT; ++kt) {
        int k0 = kt * 32 + hi * 8;
        v16bf a0 = load_a_bf16(arow0, k0);
        v16bf a1 = load_a_bf16(arow1, k0);
        v16bf b0 = load_b_sw(woutsw, btile0 + 0 * KT + kt, lane);
        v16bf b1 = load_b_sw(woutsw, btile0 + 1 * KT + kt, lane);
        v16bf b2 = load_b_sw(woutsw, btile0 + 2 * KT + kt, lane);
        v16bf b3 = load_b_sw(woutsw, btile0 + 3 * KT + kt, lane);
        c[0][0] = wmma_bf16(a0, b0, c[0][0]);
        c[1][0] = wmma_bf16(a1, b0, c[1][0]);
        c[0][1] = wmma_bf16(a0, b1, c[0][1]);
        c[1][1] = wmma_bf16(a1, b1, c[1][1]);
        c[0][2] = wmma_bf16(a0, b2, c[0][2]);
        c[1][2] = wmma_bf16(a1, b2, c[1][2]);
        c[0][3] = wmma_bf16(a0, b3, c[0][3]);
        c[1][3] = wmma_bf16(a1, b3, c[1][3]);
    }

#pragma unroll
    for (int i = 0; i < 4; ++i) {
        int n = (nb * 4 + i) * 16 + (lane & 15);
        float bv = bias[n];
#pragma unroll
        for (int m = 0; m < 2; ++m) {
            int tt = mb * 2 + m;                     // timestep
#pragma unroll
            for (int r = 0; r < 8; ++r) {
                int bb = r + hi * 8;                 // batch
                out[((long long)bb * T_ + tt) * V_ + n] = c[m][i][r] + bv;
            }
        }
    }
}

// Variant B: on-the-fly fp32 -> bf16 conversion of W_out (workspace too small)
__global__ void gemm_out_raw_kernel(const __bf16* __restrict__ hids, const float* __restrict__ wout,
                                    const float* __restrict__ bias, float* __restrict__ out) {
    constexpr int NT = V_ / 16;
    constexpr int KT = H_ / 32;
    long long wave = (long long)blockIdx.x * (blockDim.x >> 5) + (threadIdx.x >> 5);
    int lane = threadIdx.x & 31;
    int mt = (int)(wave / NT);
    int nt = (int)(wave % NT);
    int hi = lane >> 4;
    const __bf16* arow = hids + (long long)(mt * 16 + (lane & 15)) * H_;
    int n = nt * 16 + (lane & 15);
    const float* wcol = wout + n;
    v8f c = {};
    for (int kt = 0; kt < KT; ++kt) {
        v16bf a = load_a_bf16(arow, kt * 32 + hi * 8);
        int kb = kt * 32 + hi * 8;
        v16bf b;
#pragma unroll
        for (int j = 0; j < 16; ++j)
            b[j] = (__bf16)wcol[(long long)(kb + (j & 7) + ((j >> 3) << 4)) * V_];
        c = wmma_bf16(a, b, c);
    }
    float bv = bias[n];
#pragma unroll
    for (int r = 0; r < 8; ++r) {
        int bb = r + hi * 8;
        out[((long long)bb * T_ + mt) * V_ + n] = c[r] + bv;
    }
}

// ---------------- host launcher ----------------
extern "C" void kernel_launch(void* const* d_in, const int* in_sizes, int n_in,
                              void* d_out, int out_size, void* d_ws, size_t ws_size,
                              hipStream_t stream) {
    (void)in_sizes; (void)n_in; (void)out_size;
    const float* x     = (const float*)d_in[0];   // [B,T,E]
    const float* noise = (const float*)d_in[1];   // [B,T,H]
    const float* win   = (const float*)d_in[2];   // [E,H]
    const float* wrec  = (const float*)d_in[3];   // [H,H]
    const float* wout  = (const float*)d_in[4];   // [H,V]
    const float* bias  = (const float*)d_in[5];   // [V]
    float* out = (float*)d_out;

    // workspace layout (bytes)
    const size_t SZ_WIN  = (size_t)E_ * H_ * 2;        //   2 MB
    const size_t SZ_WREC = (size_t)H_ * H_ * 2;        // 8.4 MB
    const size_t SZ_INP  = (size_t)B_ * T_ * H_ * 4;   //  33 MB
    const size_t SZ_MEM  = (size_t)B_ * H_ * 4;
    const size_t SZ_H    = (size_t)B_ * H_ * 2;
    const size_t SZ_HIDS = (size_t)T_ * B_ * H_ * 2;   //  17 MB
    const size_t SZ_WOUT = (size_t)H_ * V_ * 2;        // 131 MB (optional)

    char* ws = (char*)d_ws;
    size_t off = 0;
    auto take = [&](size_t sz) { char* p = ws + off; off += (sz + 255) & ~(size_t)255; return p; };
    __bf16* winsw  = (__bf16*)take(SZ_WIN);
    __bf16* wrecsw = (__bf16*)take(SZ_WREC);
    float*  inp    = (float*) take(SZ_INP);
    float*  mem    = (float*) take(SZ_MEM);
    __bf16* hA     = (__bf16*)take(SZ_H);
    __bf16* hB     = (__bf16*)take(SZ_H);
    __bf16* hids   = (__bf16*)take(SZ_HIDS);
    size_t base_end = off;
    __bf16* woutsw = (__bf16*)take(SZ_WOUT);
    bool useSwizzledWout = (ws_size >= base_end + SZ_WOUT);

    // 0) zero LIF state
    {
        int n = B_ * H_;
        zero_state_kernel<<<(n + 255) / 256, 256, 0, stream>>>(mem, hA);
    }
    // 1) weight conversions / swizzles
    {
        long long n = (long long)E_ * H_;
        swizzleB_kernel<<<(unsigned)((n + 255) / 256), 256, 0, stream>>>(win, winsw, E_, H_, 0);
    }
    {
        long long n = (long long)H_ * H_;
        swizzleB_kernel<<<(unsigned)((n + 255) / 256), 256, 0, stream>>>(wrec, wrecsw, H_, H_, 1);
    }
    if (useSwizzledWout) {
        long long n = (long long)H_ * V_;
        swizzleB_kernel<<<(unsigned)((n + 255) / 256), 256, 0, stream>>>(wout, woutsw, H_, V_, 0);
    }
    // 2) input projection: 256 x 128 tiles, 8 waves/block
    {
        int tiles = (B_ * T_ / 16) * (H_ / 16);
        gemm_input_kernel<<<tiles / 8, 256, 0, stream>>>(x, winsw, inp);
    }
    // 3) recurrent LIF scan: one launch per timestep, double-buffered h_prev
    for (int t = 0; t < T_; ++t) {
        const __bf16* hin = (t & 1) ? hB : hA;
        __bf16*       hout = (t & 1) ? hA : hB;
        lif_step_kernel<<<(H_ / 16) / 8, 256, 0, stream>>>(hin, wrecsw, inp, noise,
                                                           mem, hout, hids, t);
    }
    // 4) output projection
    if (useSwizzledWout) {
        // 128 M-blocks (2 tiles) x 500 N-blocks (4 tiles) = 64000 waves, 8 waves/block
        int waves = ((B_ * T_) / 32) * (V_ / 64);
        gemm_out_sw_kernel<<<waves / 8, 256, 0, stream>>>(hids, woutsw, bias, out);
    } else {
        long long tiles = (long long)(B_ * T_ / 16) * (V_ / 16);
        gemm_out_raw_kernel<<<(unsigned)(tiles / 8), 256, 0, stream>>>(hids, wout, bias, out);
    }
}
